// RGCN_30434138260156
// MI455X (gfx1250) — matrix operand order repaired and checked by reading, not compiled, not run on hardware
//
#include <hip/hip_runtime.h>

// ---------------------------------------------------------------------------
// RGCN (3-layer, basis decomposition, mean aggregation) + mean pool + linear
// Target: gfx1250 (MI455X), wave32, bf16 WMMA with f32 accumulate.
// GEMM stages the shared B panel through LDS (loaded once per block, read by
// all 8 waves) with a software pipeline overlapping next-chunk global loads
// with current-chunk WMMA compute.
// ---------------------------------------------------------------------------

typedef __bf16 bf16_t;
typedef __attribute__((ext_vector_type(16))) __bf16 v16bf;
typedef __attribute__((ext_vector_type(8)))  __bf16 bf16x8;
typedef __attribute__((ext_vector_type(8)))  float  v8f;

// Problem constants (match reference)
#define NN   50000
#define EE   1600000
#define RR   16
#define HH   128
#define GG   64
#define NB   4
#define KTOT 2176          // R*H + H  (relation part + root part)
#define KT_STEPS 68        // KTOT / 32
#define CK   4             // kt steps per LDS chunk
#define NCHUNK (KT_STEPS / CK)          // 17
#define CHUNK_UNITS (CK * 8 * 32 * 2)   // bf16x8 units per chunk = 2048 (32 KB)

// ---------------------------------------------------------------------------
// Edge / batch counting (done once; identical across layers)
// ---------------------------------------------------------------------------
__global__ void count_edges_kernel(const int* __restrict__ ei,
                                   const int* __restrict__ et,
                                   int* __restrict__ cnt) {
    int e = blockIdx.x * blockDim.x + threadIdx.x;
    if (e < EE) {
        int dst = ei[EE + e];
        int t   = et[e];
        atomicAdd(&cnt[dst * RR + t], 1);
    }
}

__global__ void count_batch_kernel(const int* __restrict__ batch,
                                   int* __restrict__ gcnt) {
    int n = blockIdx.x * blockDim.x + threadIdx.x;
    if (n < NN) atomicAdd(&gcnt[batch[n]], 1);
}

// ---------------------------------------------------------------------------
// Per-edge scatter: agg[(dst*R+type)*128 + f] += x[src*128 + f]
// ---------------------------------------------------------------------------
__global__ void scatter_kernel(const float* __restrict__ x,
                               const int* __restrict__ ei,
                               const int* __restrict__ et,
                               float* __restrict__ agg) {
    long tid = (long)blockIdx.x * blockDim.x + threadIdx.x;
    if (tid >= (long)EE * HH) return;
    int e = (int)(tid >> 7);
    int f = (int)(tid & 127);
    int src = ei[e];
    int dst = ei[EE + e];
    int t   = et[e];
    atomicAdd(&agg[((long)(dst * RR + t)) * HH + f], x[(long)src * HH + f]);
}

// ---------------------------------------------------------------------------
// A-matrix assembly: Ab[n, r*128+k] = bf16( agg / max(cnt,1) ),
//                    Ab[n, 2048+k]  = bf16( x_in[n,k] )
// ---------------------------------------------------------------------------
__global__ void build_a_mean_kernel(const float* __restrict__ agg,
                                    const int* __restrict__ cnt,
                                    bf16_t* __restrict__ Ab) {
    long tid = (long)blockIdx.x * blockDim.x + threadIdx.x;
    if (tid >= (long)NN * (RR * HH)) return;
    int n  = (int)(tid / (RR * HH));
    int rk = (int)(tid % (RR * HH));
    int c  = cnt[n * RR + (rk >> 7)];
    float denom = (c > 0) ? (float)c : 1.0f;
    Ab[(long)n * KTOT + rk] = (bf16_t)(agg[tid] / denom);
}

__global__ void build_a_x_kernel(const float* __restrict__ xin,
                                 bf16_t* __restrict__ Ab) {
    long tid = (long)blockIdx.x * blockDim.x + threadIdx.x;
    if (tid >= (long)NN * HH) return;
    int n = (int)(tid >> 7);
    int k = (int)(tid & 127);
    Ab[(long)n * KTOT + (RR * HH) + k] = (bf16_t)xin[tid];
}

// ---------------------------------------------------------------------------
// Weight stack: W[k,o], k<2048 -> sum_b comp[r,b]*bases[b,i,o]; k>=2048 -> root
// ---------------------------------------------------------------------------
__global__ void build_w_kernel(const float* __restrict__ bases,
                               const float* __restrict__ comp,
                               const float* __restrict__ root,
                               float* __restrict__ W) {
    int tid = blockIdx.x * blockDim.x + threadIdx.x;
    if (tid >= KTOT * HH) return;
    int k = tid >> 7;
    int o = tid & 127;
    float v;
    if (k < RR * HH) {
        int r = k >> 7;
        int i = k & 127;
        v = 0.0f;
        #pragma unroll
        for (int b = 0; b < NB; ++b)
            v += comp[r * NB + b] * bases[((long)b * HH + i) * HH + o];
    } else {
        v = root[(long)(k - RR * HH) * HH + o];
    }
    W[tid] = v;
}

// Pack B into this file's WMMA B fragment order (16-bit B 32x16, same k order as A):
// lane L -> column n = L%16, K = (L/16)*16 + j, j = 0..15 contiguous per lane.
__global__ void pack_b_kernel(const float* __restrict__ W,
                              bf16_t* __restrict__ Bp) {
    int tid = blockIdx.x * blockDim.x + threadIdx.x;
    if (tid >= KT_STEPS * 8 * 32 * 16) return;
    int j    = tid & 15;
    int lane = (tid >> 4) & 31;
    int nt   = (tid >> 9) & 7;
    int kt   = tid >> 12;
    int k = kt * 32 + (lane >> 4) * 16 + j;
    int c = nt * 16 + (lane & 15);
    Bp[tid] = (bf16_t)W[k * HH + c];
}

// ---------------------------------------------------------------------------
// WMMA GEMM: out[N,128] = Ab[N,2176] x Bpack[2176,128] + bias (+ReLU)
// 256 threads = 8 waves; each wave owns a 16-node tile and the full 128-wide
// output (8 accumulators of 16x16 f32). B chunks (32 KB = 4 kt steps) are
// staged once per block into LDS; waves feed WMMA from ds_load_b128 while the
// next chunk's global loads are already in flight (software pipeline).
// ---------------------------------------------------------------------------
__global__ __launch_bounds__(256)
void rgcn_gemm_kernel(const bf16_t* __restrict__ A,
                      const bf16_t* __restrict__ Bp,
                      const float* __restrict__ bias,
                      float* __restrict__ out,
                      int do_relu) {
    __shared__ bf16x8 sB[CHUNK_UNITS];           // 32 KB B chunk

    const int tid  = threadIdx.x;
    const int lane = tid & 31;
    const int wave = tid >> 5;
    const int node_base = (blockIdx.x * 8 + wave) * 16;

    const int m    = lane & 15;   // row (A) / column (B,C) within tile
    const int half = lane >> 4;   // lane-half selector

    v8f acc[8];
    #pragma unroll
    for (int t = 0; t < 8; ++t)
        #pragma unroll
        for (int i = 0; i < 8; ++i) acc[t][i] = 0.0f;

    int row = node_base + m;
    if (row >= NN) row = NN - 1;              // clamp (EXEC must stay full)
    const bf16_t* arow = A + (long)row * KTOT;
    const int kA = half * 16;                 // A-fragment K sub-offset

    const bf16x8* gB = (const bf16x8*)Bp;     // global B in 16-byte units
    bf16x8 stage[8];                          // 128 B of staged B per thread

    // preload chunk 0 into registers
    #pragma unroll
    for (int i = 0; i < 8; ++i) stage[i] = gB[(long)tid * 8 + i];

    for (int c = 0; c < NCHUNK; ++c) {
        __syncthreads();                       // prior chunk's readers done
        #pragma unroll
        for (int i = 0; i < 8; ++i) sB[tid * 8 + i] = stage[i];
        __syncthreads();                       // chunk visible to all waves

        if (c + 1 < NCHUNK) {                  // kick off next chunk's loads
            #pragma unroll
            for (int i = 0; i < 8; ++i)
                stage[i] = gB[(long)(c + 1) * CHUNK_UNITS + (long)tid * 8 + i];
        }

        #pragma unroll
        for (int kl = 0; kl < CK; ++kl) {
            const int k0 = (c * CK + kl) * 32;
            // A fragment: VGPRs 0-3 <- K = k0+kA+{0..7}; VGPRs 4-7 <- +8
            bf16x8 alo = *(const bf16x8*)(arow + k0 + kA);
            bf16x8 ahi = *(const bf16x8*)(arow + k0 + kA + 8);
            __builtin_prefetch(arow + k0 + 128 + kA, 0, 3);  // global_prefetch_b8
            v16bf av;
            #pragma unroll
            for (int i = 0; i < 8; ++i) { av[i] = alo[i]; av[i + 8] = ahi[i]; }

            #pragma unroll
            for (int nt = 0; nt < 8; ++nt) {
                const int u = ((kl * 8 + nt) * 32 + lane) * 2;
                bf16x8 blo = sB[u];            // ds_load_b128
                bf16x8 bhi = sB[u + 1];        // ds_load_b128
                v16bf bv;
                #pragma unroll
                for (int i = 0; i < 8; ++i) { bv[i] = blo[i]; bv[i + 8] = bhi[i]; }
                acc[nt] = __builtin_amdgcn_wmma_f32_16x16x32_bf16(
                    false, av, false, bv, (short)0, acc[nt], false, false);
            }
        }
    }

    // C layout: lane half selects M offset (+8); col = lane%16
    #pragma unroll
    for (int nt = 0; nt < 8; ++nt) {
        const int col = nt * 16 + m;
        const float b = bias[col];
        #pragma unroll
        for (int v = 0; v < 8; ++v) {
            int mrow = node_base + v + half * 8;
            if (mrow < NN) {
                float val = acc[nt][v] + b;
                if (do_relu) val = fmaxf(val, 0.0f);
                out[(long)mrow * HH + col] = val;
            }
        }
    }
}

// ---------------------------------------------------------------------------
// Global mean pool (sum part) and final linear head
// ---------------------------------------------------------------------------
__global__ void pool_sum_kernel(const float* __restrict__ h,
                                const int* __restrict__ batch,
                                float* __restrict__ pool) {
    long tid = (long)blockIdx.x * blockDim.x + threadIdx.x;
    if (tid >= (long)NN * HH) return;
    int n = (int)(tid >> 7);
    int f = (int)(tid & 127);
    atomicAdd(&pool[batch[n] * HH + f], h[tid]);
}

__global__ void head_kernel(const float* __restrict__ pool,
                            const int* __restrict__ gcnt,
                            const int* __restrict__ rel_labels,
                            const float* __restrict__ rel_emb,
                            const float* __restrict__ lin_w,
                            const float* __restrict__ lin_b,
                            float* __restrict__ out) {
    int t = threadIdx.x;
    if (t >= GG * 2) return;
    int g = t >> 1;
    int c = t & 1;
    float denom = (gcnt[g] > 0) ? (float)gcnt[g] : 1.0f;
    const float* re = rel_emb + (long)rel_labels[g] * HH;
    float s = lin_b[c];
    for (int k = 0; k < HH; ++k) {
        s += (pool[g * HH + k] / denom) * lin_w[k * 2 + c];
        s += re[k] * lin_w[(HH + k) * 2 + c];
    }
    out[g * 2 + c] = s;
}

// ---------------------------------------------------------------------------
// Launch
// ---------------------------------------------------------------------------
extern "C" void kernel_launch(void* const* d_in, const int* in_sizes, int n_in,
                              void* d_out, int out_size, void* d_ws, size_t ws_size,
                              hipStream_t stream) {
    const float* x          = (const float*)d_in[0];
    const int*   ei         = (const int*)d_in[1];
    const int*   et         = (const int*)d_in[2];
    const int*   batch      = (const int*)d_in[3];
    const int*   rel_labels = (const int*)d_in[4];
    // d_in[5] = drop_prob (identity in eval mode)
    const float* rel_emb    = (const float*)d_in[6];
    const float* bases[3] = {(const float*)d_in[7],  (const float*)d_in[11], (const float*)d_in[15]};
    const float* comp[3]  = {(const float*)d_in[8],  (const float*)d_in[12], (const float*)d_in[16]};
    const float* root[3]  = {(const float*)d_in[9],  (const float*)d_in[13], (const float*)d_in[17]};
    const float* bias[3]  = {(const float*)d_in[10], (const float*)d_in[14], (const float*)d_in[18]};
    const float* lin_w    = (const float*)d_in[19];
    const float* lin_b    = (const float*)d_in[20];
    float* out = (float*)d_out;

    // --- carve workspace (256B-aligned chunks) ---
    char* base = (char*)d_ws;
    size_t off = 0;
    auto alloc = [&](size_t bytes) -> void* {
        void* p = base + off;
        off += (bytes + 255) & ~(size_t)255;
        return p;
    };
    size_t aggBytes = (size_t)NN * RR * HH * sizeof(float);        // 409.6 MB
    float*  agg   = (float*)alloc(aggBytes);
    int*    cnt   = (int*)  alloc((size_t)NN * RR * sizeof(int));
    int*    gcnt  = (int*)  alloc((size_t)GG * sizeof(int));
    bf16_t* Ab    = (bf16_t*)alloc((size_t)NN * KTOT * sizeof(bf16_t)); // 217.6 MB
    float*  Wfull = (float*)alloc((size_t)KTOT * HH * sizeof(float));
    bf16_t* Bp    = (bf16_t*)alloc((size_t)KTOT * HH * sizeof(bf16_t));
    float*  h1    = (float*)alloc((size_t)NN * HH * sizeof(float));
    float*  h2    = (float*)alloc((size_t)NN * HH * sizeof(float));
    float*  h3    = (float*)alloc((size_t)NN * HH * sizeof(float));
    float*  pool  = (float*)alloc((size_t)GG * HH * sizeof(float));

    const int TPB = 256;
    const long scatterTotal = (long)EE * HH;
    const int  scatterBlocks = (int)((scatterTotal + TPB - 1) / TPB);
    const int  meanBlocks    = (int)(((long)NN * RR * HH + TPB - 1) / TPB);
    const int  nhBlocks      = (int)(((long)NN * HH + TPB - 1) / TPB);
    const int  wBlocks       = (KTOT * HH + TPB - 1) / TPB;
    const int  pBlocks       = (KT_STEPS * 8 * 32 * 16 + TPB - 1) / TPB;
    const int  gemmBlocks    = (NN + 127) / 128;

    // per-relation edge counts + per-graph node counts (layer-invariant)
    hipMemsetAsync(cnt,  0, (size_t)NN * RR * sizeof(int), stream);
    hipMemsetAsync(gcnt, 0, (size_t)GG * sizeof(int), stream);
    count_edges_kernel<<<(EE + TPB - 1) / TPB, TPB, 0, stream>>>(ei, et, cnt);
    count_batch_kernel<<<(NN + TPB - 1) / TPB, TPB, 0, stream>>>(batch, gcnt);

    const float* cur = x;
    float* hout[3] = {h1, h2, h3};
    for (int l = 0; l < 3; ++l) {
        hipMemsetAsync(agg, 0, aggBytes, stream);
        scatter_kernel<<<scatterBlocks, TPB, 0, stream>>>(cur, ei, et, agg);
        build_a_mean_kernel<<<meanBlocks, TPB, 0, stream>>>(agg, cnt, Ab);
        build_a_x_kernel<<<nhBlocks, TPB, 0, stream>>>(cur, Ab);
        build_w_kernel<<<wBlocks, TPB, 0, stream>>>(bases[l], comp[l], root[l], Wfull);
        pack_b_kernel<<<pBlocks, TPB, 0, stream>>>(Wfull, Bp);
        rgcn_gemm_kernel<<<gemmBlocks, TPB, 0, stream>>>(Ab, Bp, bias[l], hout[l],
                                                         (l < 2) ? 1 : 0);
        cur = hout[l];
    }

    hipMemsetAsync(pool, 0, (size_t)GG * HH * sizeof(float), stream);
    pool_sum_kernel<<<nhBlocks, TPB, 0, stream>>>(h3, batch, pool);
    head_kernel<<<1, 128, 0, stream>>>(pool, gcnt, rel_labels, rel_emb,
                                       lin_w, lin_b, out);
}
